// VectorQuantizer_49615462203424
// MI455X (gfx1250) — compile-verified, matrix operand order checked by reading, not run on hardware
//
#include <hip/hip_runtime.h>
#include <hip/hip_bf16.h>
#include <stdint.h>

typedef __attribute__((ext_vector_type(16))) _Float16 v16h;
typedef __attribute__((ext_vector_type(8)))  _Float16 v8h;
typedef __attribute__((ext_vector_type(8)))  float    v8f;
typedef __attribute__((ext_vector_type(4)))  float    v4f;

#define K_EMB 1024
#define D_EMB 256
#define N_ROWS 32768

union V16 { v16h v; v8h h[2]; };

// ---------------------------------------------------------------------------
// Zero-fill the one-hot encodings region. Base is 4B-aligned only (odd float
// offset in d_out), so peel a head/tail and stream the interior with
// non-temporal b128 stores (134 MB is the largest traffic component; NT keeps
// it from displacing the L2-resident f16 codebook).
// ---------------------------------------------------------------------------
__global__ void vq_zero(float* __restrict__ p, long n) {
    const long i      = (long)blockIdx.x * blockDim.x + threadIdx.x;
    const long stride = (long)gridDim.x * blockDim.x;

    long head = (long)(((16u - ((uintptr_t)p & 15u)) & 15u) >> 2);  // floats to 16B
    if (head > n) head = n;
    if (i < head) p[i] = 0.0f;

    const long n4 = (n - head) >> 2;
    v4f* __restrict__ p4 = (v4f*)(p + head);
    const v4f z = {0.0f, 0.0f, 0.0f, 0.0f};
    for (long j = i; j < n4; j += stride)
        __builtin_nontemporal_store(z, &p4[j]);

    const long tail = head + (n4 << 2) + i;
    if (tail < n) p[tail] = 0.0f;
}

// ---------------------------------------------------------------------------
// Codebook f32 -> f16 staging, e_sq per code, zero histogram / mse accum.
// One block per code row; 256 threads = one per dim.
// ---------------------------------------------------------------------------
__global__ __launch_bounds__(256)
void vq_prep(const float* __restrict__ cb, _Float16* __restrict__ cb16,
             float* __restrict__ esq, unsigned int* __restrict__ hist,
             float* __restrict__ mse_accum)
{
    __shared__ float sRed[8];
    const int k = blockIdx.x;
    const int d = threadIdx.x;
    float v = cb[(size_t)k * D_EMB + d];
    cb16[(size_t)k * D_EMB + d] = (_Float16)v;
    float p = v * v;
    for (int off = 1; off < 32; off <<= 1) p += __shfl_xor(p, off, 32);
    const int wave = d >> 5, lane = d & 31;
    if (lane == 0) sRed[wave] = p;
    __syncthreads();
    if (d == 0) {
        float s = 0.0f;
        for (int w = 0; w < 8; ++w) s += sRed[w];
        esq[k] = s;
        hist[k] = 0u;
        if (k == 0) *mse_accum = 0.0f;
    }
}

// ---------------------------------------------------------------------------
// Main: 1 workgroup = 16 rows of x; 8 waves each own 8 column tiles (16 codes).
// Score = e_sq - 2 * <x, e> via v_wmma_f32_16x16x32_f16, argmin over K=1024.
// ---------------------------------------------------------------------------
__global__ __launch_bounds__(256)
void vq_main(const float* __restrict__ x, const float* __restrict__ cb,
             const _Float16* __restrict__ cb16, const float* __restrict__ esq,
             float* __restrict__ out_q, float* __restrict__ out_enc,
             float* __restrict__ out_idx, float* __restrict__ mse_accum,
             unsigned int* __restrict__ hist)
{
    __shared__ __align__(16) _Float16 sX[16 * D_EMB];   // 8 KB f16 x-tile
    __shared__ float sBestS[8][16];
    __shared__ int   sBestI[8][16];
    __shared__ int   sIdx[16];
    __shared__ float sRed[8];

    const int tid  = threadIdx.x;
    const int wave = tid >> 5;
    const int lane = tid & 31;
    const int half = lane >> 4;     // 0: lanes 0-15, 1: lanes 16-31
    const int l16  = lane & 15;
    const int row0 = blockIdx.x * 16;

    // Stage 16x256 f32 x-tile into LDS as f16.
    for (int i = 0; i < 16; ++i) {
        int e = tid + i * 256;
        sX[e] = (_Float16)x[(size_t)row0 * D_EMB + e];
    }
    __syncthreads();

    // Hoist all A-fragments (K = 0..255) into registers.
    // 16-bit A layout: lane = M + 16*hi; elems 0..7 -> K = k0+8*hi+(0..7),
    //                  elems 8..15 -> K = k0+16+8*hi+(0..7).
    V16 areg[8];
    {
        const int arow = l16;
        const int off  = half * 8;
        for (int s = 0; s < 8; ++s) {
            const int k0 = s * 32;
            areg[s].h[0] = *(const v8h*)&sX[arow * D_EMB + k0 + off];
            areg[s].h[1] = *(const v8h*)&sX[arow * D_EMB + k0 + 16 + off];
        }
    }

    float bestS[8];
    int   bestI[8];
    for (int r = 0; r < 8; ++r) { bestS[r] = 3.0e38f; bestI[r] = 0; }

    for (int t = 0; t < 8; ++t) {                 // ascending tiles per wave
        const int tile    = t * 8 + wave;
        const int colbase = tile * 16;
        const int cbrow   = colbase + l16;        // B column = codebook row
        const _Float16* bp = cb16 + (size_t)cbrow * D_EMB;

        v8f c = {};
        for (int s = 0; s < 8; ++s) {
            const int k0 = s * 32;
            V16 breg;
            breg.h[0] = *(const v8h*)&bp[k0 + half * 8];
            breg.h[1] = *(const v8h*)&bp[k0 + 16 + half * 8];
            c = __builtin_amdgcn_wmma_f32_16x16x32_f16(
                    false, areg[s].v, false, breg.v, (short)0, c, false, false);
        }

        const float e2 = esq[cbrow];
        for (int r = 0; r < 8; ++r) {             // row = r + 8*half, col = cbrow
            float score = fmaf(-2.0f, c[r], e2);
            if (score < bestS[r]) { bestS[r] = score; bestI[r] = cbrow; }
        }
    }

    // Argmin across the 16 lanes of each half (columns), ties -> lower index.
    for (int r = 0; r < 8; ++r) {
        float s = bestS[r]; int ci = bestI[r];
        for (int off = 1; off < 16; off <<= 1) {
            float os = __shfl_xor(s, off, 32);
            int   oc = __shfl_xor(ci, off, 32);
            if (os < s || (os == s && oc < ci)) { s = os; ci = oc; }
        }
        bestS[r] = s; bestI[r] = ci;
    }
    if (l16 == 0) {
        for (int r = 0; r < 8; ++r) {
            sBestS[wave][r + half * 8] = bestS[r];
            sBestI[wave][r + half * 8] = bestI[r];
        }
    }
    __syncthreads();

    // Argmin across the 8 waves; emit indices / one-hot / histogram.
    if (tid < 16) {
        float s = sBestS[0][tid]; int ci = sBestI[0][tid];
        for (int w = 1; w < 8; ++w) {
            float os = sBestS[w][tid]; int oc = sBestI[w][tid];
            if (os < s || (os == s && oc < ci)) { s = os; ci = oc; }
        }
        sIdx[tid] = ci;
        const int row = row0 + tid;
        out_idx[row] = (float)ci;
        out_enc[(size_t)row * K_EMB + ci] = 1.0f;
        atomicAdd(&hist[ci], 1u);
    }
    __syncthreads();

    // Epilogue: quantized (= selected codebook row; STE forward value) + MSE.
    float part = 0.0f;
    for (int r = 0; r < 16; ++r) {
        const int row = row0 + r;
        const int ci  = sIdx[r];
        float e  = cb[(size_t)ci * D_EMB + tid];
        float xv = x[(size_t)row * D_EMB + tid];
        out_q[(size_t)row * D_EMB + tid] = e;
        float d = e - xv;
        part = fmaf(d, d, part);
    }
    for (int off = 1; off < 32; off <<= 1) part += __shfl_xor(part, off, 32);
    if (lane == 0) sRed[wave] = part;
    __syncthreads();
    if (tid == 0) {
        float sum = 0.0f;
        for (int w = 0; w < 8; ++w) sum += sRed[w];
        atomicAdd(mse_accum, sum);
    }
}

// ---------------------------------------------------------------------------
// Finalize: usage loss from histogram + total loss.
// Forward-value identity: e_latent_loss == q_latent_loss == MSE.
// ---------------------------------------------------------------------------
__global__ __launch_bounds__(1024)
void vq_final(const unsigned int* __restrict__ hist,
              const float* __restrict__ mse_accum,
              float* __restrict__ out_loss)
{
    __shared__ float sRed[32];
    const int k = threadIdx.x;
    float u = (float)hist[k] * (1.0f / (float)N_ROWS) - (1.0f / (float)K_EMB);
    float p = u * u;
    for (int off = 1; off < 32; off <<= 1) p += __shfl_xor(p, off, 32);
    const int wave = k >> 5, lane = k & 31;
    if (lane == 0) sRed[wave] = p;
    __syncthreads();
    if (k == 0) {
        float s = 0.0f;
        for (int w = 0; w < 32; ++w) s += sRed[w];
        float usage_loss = s * (1.0f / (float)K_EMB);
        float mse = *mse_accum * (1.0f / ((float)N_ROWS * (float)D_EMB));
        out_loss[0] = 1.25f * mse + 0.1f * usage_loss;   // (1 + 0.25)*mse + 0.1*usage
    }
}

// ---------------------------------------------------------------------------
extern "C" void kernel_launch(void* const* d_in, const int* in_sizes, int n_in,
                              void* d_out, int out_size, void* d_ws, size_t ws_size,
                              hipStream_t stream)
{
    const float* x  = (const float*)d_in[0];   // (16,2048,256) f32
    const float* cb = (const float*)d_in[1];   // (1024,256)    f32
    float* out = (float*)d_out;

    // Output layout (flat, return order):
    const size_t Q_OFF    = 0;                 // quantized_ste  8,388,608
    const size_t LOSS_OFF = 8388608;           // loss           1
    const size_t ENC_OFF  = 8388609;           // encodings      33,554,432
    const size_t IDX_OFF  = 41943041;          // indices        32,768

    // Workspace layout:
    _Float16*     cb16 = (_Float16*)d_ws;                            // 512 KB
    float*        esq  = (float*)((char*)d_ws + 524288);             // 4 KB
    unsigned int* hist = (unsigned int*)((char*)d_ws + 528384);      // 4 KB
    float*        mse  = (float*)((char*)d_ws + 532480);             // 4 B

    vq_zero <<<2048, 256, 0, stream>>>(out + ENC_OFF, (long)33554432);
    vq_prep <<<K_EMB, 256, 0, stream>>>(cb, cb16, esq, hist, mse);
    vq_main <<<N_ROWS / 16, 256, 0, stream>>>(x, cb, cb16, esq,
                                              out + Q_OFF, out + ENC_OFF,
                                              out + IDX_OFF, mse, hist);
    vq_final<<<1, 1024, 0, stream>>>(hist, mse, out + LOSS_OFF);
}
